// AttentionSequencePoolingLayer_5385888989707
// MI455X (gfx1250) — matrix operand order repaired and checked by reading, not compiled, not run on hardware
//
#include <hip/hip_runtime.h>
#include <hip/hip_bf16.h>
#include <math.h>

// ---------------------------------------------------------------------------
// AttentionSequencePoolingLayer for MI455X (gfx1250, wave32, WMMA)
//   B=4096, T=200, E=64, H1=64, H2=32
//   x = [q, k, q-k, q*k] (256) -> relu(W1) -> relu(W2) -> W3 -> masked softmax
//   Folded: x@W1 = q@(W1a+W1c) + k@(W1b-W1c) + (q*k)@W1d
// ---------------------------------------------------------------------------

typedef __attribute__((ext_vector_type(16))) _Float16 v16h;
typedef __attribute__((ext_vector_type(8)))  _Float16 v8h;
typedef __attribute__((ext_vector_type(8)))  float    v8f;

#define MASK_VAL (-4294967295.0f)   // -2^32 + 1
#define TT   200
#define EE   64
#define NTIL 13                      // ceil(200/16)
#define XSTR 136                     // 128 + 8 pad halfs  (272B, 16B aligned rows)
#define H1STR 72                     // 64 + 8 pad halfs
#define H2STR 40                     // 32 + 8 pad halfs

static __device__ __forceinline__ void lds_fence() {
    __asm__ volatile("s_wait_dscnt 0" ::: "memory");
}

static __device__ __forceinline__ v16h make_v16(v8h lo, v8h hi) {
    v16h r;
#pragma unroll
    for (int i = 0; i < 8; ++i) { r[i] = lo[i]; r[i + 8] = hi[i]; }
    return r;
}

// ---------------------------------------------------------------------------
// Prep kernel: pack W1' (128x64, rows [k | q*k]) and W2 (64x32) into WMMA
// B-fragment order:  idx = ((chunk*NT + ntile)*32 + lane)*16 + j
//   value = W[32*chunk + 16*(lane>>4) + j][16*ntile + (lane&15)]
// ---------------------------------------------------------------------------
__global__ void attn_prep_kernel(const float* __restrict__ W1,
                                 const float* __restrict__ W2,
                                 _Float16* __restrict__ w1frag,   // 8192 halfs
                                 _Float16* __restrict__ w2frag) { // 2048 halfs
    int i = blockIdx.x * blockDim.x + threadIdx.x;
    if (i < 8192) {
        int j  = i & 15;
        int l  = (i >> 4) & 31;
        int ct = i >> 9;            // 0..15
        int c  = ct >> 2;           // K chunk 0..3
        int t  = ct & 3;            // N tile 0..3
        int kk = 32 * c + 16 * (l >> 4) + j;   // 0..127
        int n  = 16 * t + (l & 15);            // 0..63
        float v;
        if (kk < 64)  v = W1[(64 + kk) * 64 + n] - W1[(128 + kk) * 64 + n]; // k part
        else          v = W1[(192 + (kk - 64)) * 64 + n];                   // q*k part
        w1frag[i] = (_Float16)v;
    } else if (i < 8192 + 2048) {
        int i2 = i - 8192;
        int j  = i2 & 15;
        int l  = (i2 >> 4) & 31;
        int ct = i2 >> 9;           // 0..3
        int c  = ct >> 1;           // K chunk 0..1
        int t  = ct & 1;            // N tile 0..1
        int kk = 32 * c + 16 * (l >> 4) + j;   // 0..63
        int n  = 16 * t + (l & 15);            // 0..31
        w2frag[i2] = (_Float16)W2[kk * 32 + n];
    }
}

// ---------------------------------------------------------------------------
// Main kernel: one block (128 threads = 4 waves) per batch row.
// ---------------------------------------------------------------------------
__global__ void __launch_bounds__(128)
attn_main_kernel(const float* __restrict__ query,
                 const float* __restrict__ keys,
                 const int*   __restrict__ keys_len,
                 const float* __restrict__ W1,
                 const float* __restrict__ b1,
                 const float* __restrict__ b2,
                 const float* __restrict__ W3,
                 const float* __restrict__ b3,
                 const _Float16* __restrict__ w1frag,
                 const _Float16* __restrict__ w2frag,
                 float* __restrict__ out) {
    __shared__ __align__(16) _Float16 sW1f[8192];
    __shared__ __align__(16) _Float16 sW2f[2048];
    __shared__ __align__(16) _Float16 xh [4][16 * XSTR];
    __shared__ __align__(16) _Float16 h1s[4][16 * H1STR];
    __shared__ __align__(16) _Float16 h2s[4][16 * H2STR];
    __shared__ float qs[64];
    __shared__ float bias1[64];
    __shared__ float b2s[32];
    __shared__ float w3s[32];
    __shared__ float scores[208];
    __shared__ float red[128];

    const int b   = blockIdx.x;
    const int tid = threadIdx.x;
    const int wv  = tid >> 5;
    const int ln  = tid & 31;

    // ---- block init: q, b2, w3, weight fragments -> LDS -------------------
    if (tid < 64) qs[tid] = query[(size_t)b * 64 + tid];
    if (tid < 32) { b2s[tid] = b2[tid]; w3s[tid] = W3[tid]; }
    {
        const uint32_t* w1u = (const uint32_t*)w1frag;   // 4096 dwords
        const uint32_t* w2u = (const uint32_t*)w2frag;   // 1024 dwords
        uint32_t* d1 = (uint32_t*)sW1f;
        uint32_t* d2 = (uint32_t*)sW2f;
        for (int i = tid; i < 5120; i += 128) {
            if (i < 4096) d1[i] = w1u[i];
            else          d2[i - 4096] = w2u[i - 4096];
        }
    }
    __syncthreads();

    // ---- per-batch folded bias:  bias1 = b1 + q @ (W1a + W1c) -------------
    if (tid < 64) {
        float s = b1[tid];
#pragma unroll 8
        for (int k = 0; k < 64; ++k)
            s += qs[k] * (W1[k * 64 + tid] + W1[(128 + k) * 64 + tid]);
        bias1[tid] = s;
    }
    __syncthreads();

    const int   klen = keys_len[b];
    const float b3v  = b3[0];

    _Float16* xw  = xh[wv];
    _Float16* h1w = h1s[wv];
    _Float16* h2w = h2s[wv];

    // ---- MLP over 16-token tiles (round-robin across 4 waves) -------------
    for (int tl = wv; tl < NTIL; tl += 4) {
        const int m0 = tl * 16;
        const int m  = ln & 15;
        const int ch = ln >> 4;          // which 32-wide half of E this lane stages
        const int tt = m0 + m;

        // prefetch next tile's keys for this wave
        if (tt + 64 < TT)
            __builtin_prefetch(keys + ((size_t)(b * TT + tt + 64)) * EE + ch * 32, 0, 0);

        // stage x = [k | q*k] as f16 into this wave's LDS tile
        {
            _Float16* xr = xw + m * XSTR;
            const float* kp = keys + ((size_t)(b * TT + tt)) * EE + ch * 32;
#pragma unroll 8
            for (int e = 0; e < 32; ++e) {
                float kv = (tt < TT) ? kp[e] : 0.0f;
                float qv = qs[ch * 32 + e];
                xr[ch * 32 + e]      = (_Float16)kv;
                xr[64 + ch * 32 + e] = (_Float16)(qv * kv);
            }
        }
        lds_fence();

        // ---- layer 1: (16x128) @ (128x64) via 4x4 WMMA --------------------
        v8f acc[4];
        {
            const int nlo = ln & 15;
#pragma unroll
            for (int t = 0; t < 4; ++t) {
                float bv = bias1[16 * t + nlo];
#pragma unroll
                for (int r = 0; r < 8; ++r) acc[t][r] = bv;
            }
        }
        {
            const _Float16* rowp = xw + (ln & 15) * XSTR;
            const int kb = (ln >> 4) * 16;
#pragma unroll
            for (int c = 0; c < 4; ++c) {
                v8h lo = *(const v8h*)(rowp + c * 32 + kb);
                v8h hi = *(const v8h*)(rowp + c * 32 + kb + 8);
                v16h a = make_v16(lo, hi);
#pragma unroll
                for (int t = 0; t < 4; ++t) {
                    v16h bf = *(const v16h*)(sW1f + ((c * 4 + t) * 32 + ln) * 16);
                    acc[t] = __builtin_amdgcn_wmma_f32_16x16x32_f16(
                        false, a, false, bf, (short)0, acc[t], false, false);
                }
            }
        }
        // relu -> h1 tile
        {
            const int nlo = ln & 15, mhi = (ln >> 4) * 8;
#pragma unroll
            for (int t = 0; t < 4; ++t)
#pragma unroll
                for (int r = 0; r < 8; ++r) {
                    float v = acc[t][r];
                    h1w[(r + mhi) * H1STR + 16 * t + nlo] = (_Float16)(v > 0.0f ? v : 0.0f);
                }
        }
        lds_fence();

        // ---- layer 2: (16x64) @ (64x32) via 2x2 WMMA ----------------------
        v8f a2[2];
        {
            const int nlo = ln & 15;
#pragma unroll
            for (int t = 0; t < 2; ++t) {
                float bv = b2s[16 * t + nlo];
#pragma unroll
                for (int r = 0; r < 8; ++r) a2[t][r] = bv;
            }
        }
        {
            const _Float16* rowp = h1w + (ln & 15) * H1STR;
            const int kb = (ln >> 4) * 16;
#pragma unroll
            for (int c = 0; c < 2; ++c) {
                v8h lo = *(const v8h*)(rowp + c * 32 + kb);
                v8h hi = *(const v8h*)(rowp + c * 32 + kb + 8);
                v16h a = make_v16(lo, hi);
#pragma unroll
                for (int t = 0; t < 2; ++t) {
                    v16h bf = *(const v16h*)(sW2f + ((c * 2 + t) * 32 + ln) * 16);
                    a2[t] = __builtin_amdgcn_wmma_f32_16x16x32_f16(
                        false, a, false, bf, (short)0, a2[t], false, false);
                }
            }
        }
        // relu -> h2 tile
        {
            const int nlo = ln & 15, mhi = (ln >> 4) * 8;
#pragma unroll
            for (int t = 0; t < 2; ++t)
#pragma unroll
                for (int r = 0; r < 8; ++r) {
                    float v = a2[t][r];
                    h2w[(r + mhi) * H2STR + 16 * t + nlo] = (_Float16)(v > 0.0f ? v : 0.0f);
                }
        }
        lds_fence();

        // ---- layer 3: score = h2 @ W3 + b3, masked ------------------------
        {
            const int mm = ln & 15;
            const int nb = (ln >> 4) * 16;
            const _Float16* hr = h2w + mm * H2STR + nb;
            float p = 0.0f;
#pragma unroll
            for (int j = 0; j < 16; ++j) p += (float)hr[j] * w3s[nb + j];
            p += __shfl_xor(p, 16, 32);
            const int tok = m0 + mm;
            if (ln < 16 && tok < TT) {
                float sc = p + b3v;
                scores[tok] = (tok < klen) ? sc : MASK_VAL;
            }
        }
    }
    __syncthreads();

    // ---- masked softmax over T=200 ----------------------------------------
    float lm = -3.0e38f;
    for (int t = tid; t < TT; t += 128) lm = fmaxf(lm, scores[t]);
    red[tid] = lm;
    __syncthreads();
    for (int s = 64; s > 0; s >>= 1) {
        if (tid < s) red[tid] = fmaxf(red[tid], red[tid + s]);
        __syncthreads();
    }
    const float smax = red[0];
    __syncthreads();

    float ls = 0.0f;
    for (int t = tid; t < TT; t += 128) {
        float e = __expf(scores[t] - smax);
        scores[t] = e;
        ls += e;
    }
    red[tid] = ls;
    __syncthreads();
    for (int s = 64; s > 0; s >>= 1) {
        if (tid < s) red[tid] += red[tid + s];
        __syncthreads();
    }
    const float inv = 1.0f / red[0];
    __syncthreads();

    // ---- out[b] = softmax(scores) @ keys[b] -------------------------------
    {
        const int e = tid & 63;
        const int s = tid >> 6;          // 0..1
        float acc = 0.0f;
        for (int t = s; t < TT; t += 2)
            acc += scores[t] * keys[((size_t)(b * TT + t)) * EE + e];
        red[tid] = acc;
        __syncthreads();
        if (tid < 64)
            out[(size_t)b * 64 + tid] = (red[tid] + red[tid + 64]) * inv;
    }
}

// ---------------------------------------------------------------------------
extern "C" void kernel_launch(void* const* d_in, const int* in_sizes, int n_in,
                              void* d_out, int out_size, void* d_ws, size_t ws_size,
                              hipStream_t stream) {
    const float* query    = (const float*)d_in[0];
    const float* keys     = (const float*)d_in[1];
    const int*   keys_len = (const int*)  d_in[2];
    const float* W1       = (const float*)d_in[3];
    const float* b1       = (const float*)d_in[4];
    const float* W2       = (const float*)d_in[5];
    const float* b2       = (const float*)d_in[6];
    const float* W3       = (const float*)d_in[7];
    const float* b3       = (const float*)d_in[8];

    const int B = in_sizes[1] / (TT * EE);   // 4096

    _Float16* w1frag = (_Float16*)d_ws;            // 8192 halfs = 16 KB
    _Float16* w2frag = w1frag + 8192;              // 2048 halfs =  4 KB

    attn_prep_kernel<<<40, 256, 0, stream>>>(W1, W2, w1frag, w2frag);
    attn_main_kernel<<<B, 128, 0, stream>>>(query, keys, keys_len,
                                            W1, b1, b2, W3, b3,
                                            w1frag, w2frag, (float*)d_out);
}
